// Attention_2370821947975
// MI455X (gfx1250) — compile-verified, matrix operand order checked by reading, not run on hardware
//
#include <hip/hip_runtime.h>
#include <stdint.h>

// ---------------------------------------------------------------------------
// Problem constants (from reference)
// ---------------------------------------------------------------------------
#define B_   16
#define N_   1024
#define DIM_ 512
#define H_   16
#define KD_  32
#define D_   128
#define DH_  (H_ * D_)              // 2048
#define HQKV_ (DH_ + 2 * H_ * KD_)  // 3072
#define SCALE_ 0.17677669529663687f // 32^-0.5
#define EPS_ 1e-5f

typedef __bf16 bf16_t;
typedef __attribute__((ext_vector_type(16))) __bf16 v16bf;
typedef __attribute__((ext_vector_type(8)))  __bf16 v8bf;
typedef __attribute__((ext_vector_type(8)))  float  v8f;
typedef __attribute__((ext_vector_type(4)))  unsigned int u32x4;
typedef __attribute__((ext_vector_type(8)))  int i32x8;
typedef __attribute__((ext_vector_type(4)))  int i32x4;

// ---------------------------------------------------------------------------
// WMMA helpers (wave32, 16x16x32 bf16 -> f32)
// ---------------------------------------------------------------------------
static __device__ __forceinline__ v8f wmma_bf16(v16bf a, v16bf b, v8f c) {
  return __builtin_amdgcn_wmma_f32_16x16x32_bf16(false, a, false, b,
                                                 (short)0, c, false, false);
}

// Build an A or B operand for 16x16x32 bf16 WMMA from a row-major tile.
// Lanes 0-15 hold K = 0..7 and 16..23; lanes 16-31 hold K = 8..15 and 24..31.
static __device__ __forceinline__ v16bf ld_ab(const bf16_t* tile, int ld, int lane) {
  const int r  = lane & 15;
  const int kh = (lane & 16) ? 8 : 0;
  const bf16_t* p = tile + r * ld + kh;
  v8bf lo = *reinterpret_cast<const v8bf*>(p);
  v8bf hi = *reinterpret_cast<const v8bf*>(p + 16);
  v16bf out;
#pragma unroll
  for (int i = 0; i < 8; ++i) { out[i] = lo[i]; out[i + 8] = hi[i]; }
  return out;
}

// C/D layout: lane l, vgpr r -> row = r + 8*(l>>4), col = l&15
static __device__ __forceinline__ int crow(int lane, int r) {
  return r + ((lane >> 4) << 3);
}

// ---------------------------------------------------------------------------
// Kernel 0: f32 -> bf16 cast (weights)
// ---------------------------------------------------------------------------
__global__ void cast_bf16_kernel(const float* __restrict__ in,
                                 bf16_t* __restrict__ out, int n) {
  int i = blockIdx.x * blockDim.x + threadIdx.x;
  if (i < n) out[i] = (bf16_t)in[i];
}

// ---------------------------------------------------------------------------
// Kernel 1: LayerNorm + cast. One wave per row of 512; 16 elements per lane.
// ---------------------------------------------------------------------------
__global__ void layernorm_kernel(const float* __restrict__ x,
                                 const float* __restrict__ gamma,
                                 const float* __restrict__ beta,
                                 bf16_t* __restrict__ xn) {
  const int lane = threadIdx.x & 31;
  const int wave = threadIdx.x >> 5;
  const size_t row = (size_t)blockIdx.x * 8 + wave;
  const float* xr = x + row * DIM_;
  const float4* xr4 = reinterpret_cast<const float4*>(xr);

  float4 v[4];
  float s = 0.f;
#pragma unroll
  for (int i = 0; i < 4; ++i) {
    v[i] = xr4[lane * 4 + i];
    s += v[i].x + v[i].y + v[i].z + v[i].w;
  }
#pragma unroll
  for (int off = 16; off >= 1; off >>= 1) s += __shfl_xor(s, off);
  const float mu = s * (1.f / DIM_);

  float vs = 0.f;
#pragma unroll
  for (int i = 0; i < 4; ++i) {
    float dx = v[i].x - mu, dy = v[i].y - mu, dz = v[i].z - mu, dw = v[i].w - mu;
    vs += dx * dx + dy * dy + dz * dz + dw * dw;
  }
#pragma unroll
  for (int off = 16; off >= 1; off >>= 1) vs += __shfl_xor(vs, off);
  const float rstd = rsqrtf(vs * (1.f / DIM_) + EPS_);

  bf16_t* xo = xn + row * DIM_ + lane * 16;
#pragma unroll
  for (int i = 0; i < 4; ++i) {
    const int e = lane * 16 + i * 4;
    float fx[4] = {v[i].x, v[i].y, v[i].z, v[i].w};
#pragma unroll
    for (int j = 0; j < 4; ++j)
      xo[i * 4 + j] = (bf16_t)((fx[j] - mu) * rstd * gamma[e + j] + beta[e + j]);
  }
}

// ---------------------------------------------------------------------------
// GEMM building blocks: wave tile = 32(M) x 64(N), double-buffered k-loop.
// ---------------------------------------------------------------------------
static __device__ __forceinline__ void gemm_load(const bf16_t* A, const bf16_t* W,
                                                 int ld, int kt, int lane,
                                                 v16bf& a0, v16bf& a1, v16bf b[4]) {
  a0 = ld_ab(A + kt * 32, ld, lane);
  a1 = ld_ab(A + (size_t)16 * ld + kt * 32, ld, lane);
#pragma unroll
  for (int cg = 0; cg < 4; ++cg)
    b[cg] = ld_ab(W + (size_t)cg * 16 * ld + kt * 32, ld, lane);
}

static __device__ __forceinline__ void gemm_mma(v8f acc[2][4],
                                                const v16bf& a0, const v16bf& a1,
                                                const v16bf b[4]) {
#pragma unroll
  for (int cg = 0; cg < 4; ++cg) {
    acc[0][cg] = wmma_bf16(a0, b[cg], acc[0][cg]);
    acc[1][cg] = wmma_bf16(a1, b[cg], acc[1][cg]);
  }
}

// ---------------------------------------------------------------------------
// Kernel 2: QKV GEMM.  qkv = xn(16384x512) @ qkv_w^T + qkv_b
// Block = 4 waves, block tile 128x64, wave tile 32x64.
// Epilogue scatters into head-major q (pre-scaled), k, and transposed v.
// ---------------------------------------------------------------------------
__global__ void qkv_gemm_kernel(const bf16_t* __restrict__ xn,
                                const bf16_t* __restrict__ qw,
                                const float* __restrict__ qb,
                                bf16_t* __restrict__ qo,
                                bf16_t* __restrict__ ko,
                                bf16_t* __restrict__ vT) {
  const int lane = threadIdx.x & 31;
  const int wave = threadIdx.x >> 5;
  const int row0 = blockIdx.x * 128 + wave * 32;
  const int col0 = blockIdx.y * 64;
  const int KT = DIM_ / 32;   // 16

  const bf16_t* A = xn + (size_t)row0 * DIM_;
  const bf16_t* W = qw + (size_t)col0 * DIM_;
  v8f acc[2][4] = {};

  v16bf a0A, a1A, bA[4], a0B, a1B, bB[4];
  gemm_load(A, W, DIM_, 0, lane, a0A, a1A, bA);
  for (int kt = 0; kt < KT; kt += 2) {
    gemm_load(A, W, DIM_, kt + 1, lane, a0B, a1B, bB);
    if (kt + 4 < KT) {
      __builtin_prefetch(A + (kt + 4) * 32, 0, 1);
      __builtin_prefetch(W + (kt + 4) * 32, 0, 1);
    }
    gemm_mma(acc, a0A, a1A, bA);
    if (kt + 2 < KT) gemm_load(A, W, DIM_, kt + 2, lane, a0A, a1A, bA);
    gemm_mma(acc, a0B, a1B, bB);
  }

#pragma unroll
  for (int m = 0; m < 2; ++m) {
    const int rowb = row0 + m * 16;
#pragma unroll
    for (int cg = 0; cg < 4; ++cg) {
      const int j = col0 + cg * 16 + (lane & 15);
      const float bj = qb[j];
      const int hh = j / (2 * KD_ + D_);
      const int c  = j % (2 * KD_ + D_);
#pragma unroll
      for (int r = 0; r < 8; ++r) {
        const int i  = rowb + crow(lane, r);
        const int bb = i >> 10;
        const int nn = i & 1023;
        const float val = acc[m][cg][r] + bj;
        if (c < KD_) {
          qo[((size_t)(bb * H_ + hh) * N_ + nn) * KD_ + c] = (bf16_t)(val * SCALE_);
        } else if (c < 2 * KD_) {
          ko[((size_t)(bb * H_ + hh) * N_ + nn) * KD_ + (c - KD_)] = (bf16_t)val;
        } else {
          vT[((size_t)(bb * H_ + hh) * D_ + (c - 2 * KD_)) * N_ + nn] = (bf16_t)val;
        }
      }
    }
  }
}

// ---------------------------------------------------------------------------
// Kernel 3: fused attention for one (b, h, 64-row block).
// K matrix (1024x32 bf16) is staged into LDS once per workgroup via the
// Tensor Data Mover (padded to an 80-byte row stride for bank spread).
// Each wave owns 16 query rows and a 16x1024 bf16 LDS score strip.
// ---------------------------------------------------------------------------
#define PSTRIDE 1040           // score-strip row stride (bf16 elems)
#define KLDS_STRIDE 40         // staged-K row stride (bf16 elems) = 64B + 16B pad
#define KLDS_ELEMS (N_ * KLDS_STRIDE)

__global__ void attn_kernel(const bf16_t* __restrict__ q,
                            const bf16_t* __restrict__ k,
                            const bf16_t* __restrict__ vT,
                            const float* __restrict__ ab,
                            const int* __restrict__ idxs,
                            bf16_t* __restrict__ aout) {
  extern __shared__ bf16_t smem[];
  const int lane = threadIdx.x & 31;
  const int wave = threadIdx.x >> 5;
  const int h = blockIdx.y;
  const int b = blockIdx.z;
  const int qrow0 = blockIdx.x * 64 + wave * 16;

  bf16_t* kLDS = smem;                                   // 81,920 B
  bf16_t* S = smem + KLDS_ELEMS + (size_t)wave * 16 * PSTRIDE;

  const bf16_t* qbase = q + ((size_t)(b * H_ + h) * N_ + qrow0) * KD_;
  const bf16_t* kbase = k + (size_t)(b * H_ + h) * N_ * KD_;
  const bf16_t* vbase = vT + (size_t)(b * H_ + h) * D_ * N_;

  // ---- Stage K into LDS (TDM) -----------------------------------------
#if __has_builtin(__builtin_amdgcn_tensor_load_to_lds)
  if (wave == 0) {
    const uint64_t ga = (uint64_t)(uintptr_t)kbase;
    const uint32_t la = (uint32_t)(uintptr_t)kLDS;
    u32x4 g0;
    g0[0] = 1u;                                    // count=1, user descriptor
    g0[1] = la;                                    // lds_addr
    g0[2] = (uint32_t)ga;                          // global_addr[31:0]
    g0[3] = (uint32_t)((ga >> 32) & 0x01FFFFFFu) | (2u << 30); // addr hi, type=2
    i32x8 g1;
    g1[0] = (1 << 16) | (1 << 20) | (3 << 22) | (3 << 25);
             // data_size=2B, pad_enable, pad every 64B, pad 4 DWORDs (16B)
    g1[1] = (int)(32u << 16);                      // tensor_dim0 = 32
    g1[2] = (int)(1024u << 16);                    // tensor_dim1 = 1024 (lo16)
    g1[3] = (int)(32u << 16);                      // tile_dim0 = 32
    g1[4] = 1024;                                  // tile_dim1 = 1024
    g1[5] = 32;                                    // tensor_dim0_stride = 32
    g1[6] = 0;
    g1[7] = 0;
    i32x4 z4 = {0, 0, 0, 0};
    i32x8 z8 = {0, 0, 0, 0, 0, 0, 0, 0};
    // 6-arg form (clang-23 / therock-10.0 signature)
    __builtin_amdgcn_tensor_load_to_lds(g0, g1, z4, z4, z8, 0);
#if __has_builtin(__builtin_amdgcn_s_wait_tensorcnt)
    __builtin_amdgcn_s_wait_tensorcnt(0);
#else
    asm volatile("s_wait_tensorcnt 0x0" ::: "memory");
#endif
  }
#else
  // Fallback: cooperative copy, 128 threads
  for (int r = threadIdx.x; r < N_; r += 128) {
    const v8bf* src = reinterpret_cast<const v8bf*>(kbase + (size_t)r * KD_);
    v8bf* dst = reinterpret_cast<v8bf*>(kLDS + (size_t)r * KLDS_STRIDE);
    dst[0] = src[0]; dst[1] = src[1]; dst[2] = src[2]; dst[3] = src[3];
  }
#endif
  __syncthreads();

  // ---- Phase 1: S = (q*scale) @ k^T  (K from LDS, double-buffered) -----
  const v16bf aQ = ld_ab(qbase, KD_, lane);
  {
    v16bf bK0 = ld_ab(kLDS, KLDS_STRIDE, lane);
    for (int mt = 0; mt < N_ / 16; mt += 2) {
      v16bf bK1 = ld_ab(kLDS + (size_t)(mt + 1) * 16 * KLDS_STRIDE, KLDS_STRIDE, lane);
      v8f c0 = {};
      c0 = wmma_bf16(aQ, bK0, c0);
      {
        const int col = mt * 16 + (lane & 15);
#pragma unroll
        for (int r = 0; r < 8; ++r)
          S[crow(lane, r) * PSTRIDE + col] = (bf16_t)c0[r];
      }
      if (mt + 2 < N_ / 16)
        bK0 = ld_ab(kLDS + (size_t)(mt + 2) * 16 * KLDS_STRIDE, KLDS_STRIDE, lane);
      v8f c1 = {};
      c1 = wmma_bf16(aQ, bK1, c1);
      {
        const int col = (mt + 1) * 16 + (lane & 15);
#pragma unroll
        for (int r = 0; r < 8; ++r)
          S[crow(lane, r) * PSTRIDE + col] = (bf16_t)c1[r];
      }
    }
  }

  // ---- Phase 2: softmax (bias folded in), 8-wide chunks ----------------
  const int row  = lane & 15;
  const int half = lane >> 4;
  bf16_t* Srow = S + row * PSTRIDE + half * 512;
  const int* idxrow = idxs + (size_t)(qrow0 + row) * N_ + half * 512;
  const float* abh = ab + h * N_;   // RES*RES == 1024

  float mx = -3.0e38f;
  for (int c0 = 0; c0 < 512; c0 += 8) {
    v8bf sv = *reinterpret_cast<v8bf*>(Srow + c0);
    const int* ip = idxrow + c0;
#pragma unroll
    for (int j = 0; j < 8; ++j) {
      const float vv = (float)sv[j] + abh[ip[j]];
      sv[j] = (bf16_t)vv;
      mx = fmaxf(mx, vv);
    }
    *reinterpret_cast<v8bf*>(Srow + c0) = sv;
  }
  mx = fmaxf(mx, __shfl_xor(mx, 16));

  float sum = 0.f;
  for (int c0 = 0; c0 < 512; c0 += 8) {
    v8bf sv = *reinterpret_cast<v8bf*>(Srow + c0);
#pragma unroll
    for (int j = 0; j < 8; ++j) {
      const float p = __expf((float)sv[j] - mx);
      sv[j] = (bf16_t)p;
      sum += p;
    }
    *reinterpret_cast<v8bf*>(Srow + c0) = sv;
  }
  sum += __shfl_xor(sum, 16);
  const float rsum = sum;

  // ---- Phase 3: O = P @ V (double-buffered) ----------------------------
  v8f o[8] = {};
  {
    v16bf aP0 = ld_ab(S, PSTRIDE, lane);
    v16bf bV0[8];
#pragma unroll
    for (int cg = 0; cg < 8; ++cg)
      bV0[cg] = ld_ab(vbase + (size_t)cg * 16 * N_, N_, lane);

    for (int kt = 0; kt < N_ / 32; kt += 2) {
      v16bf aP1 = ld_ab(S + (kt + 1) * 32, PSTRIDE, lane);
      v16bf bV1[8];
#pragma unroll
      for (int cg = 0; cg < 8; ++cg)
        bV1[cg] = ld_ab(vbase + (size_t)cg * 16 * N_ + (kt + 1) * 32, N_, lane);

#pragma unroll
      for (int cg = 0; cg < 8; ++cg)
        o[cg] = wmma_bf16(aP0, bV0[cg], o[cg]);

      if (kt + 2 < N_ / 32) {
        aP0 = ld_ab(S + (kt + 2) * 32, PSTRIDE, lane);
#pragma unroll
        for (int cg = 0; cg < 8; ++cg)
          bV0[cg] = ld_ab(vbase + (size_t)cg * 16 * N_ + (kt + 2) * 32, N_, lane);
      }
#pragma unroll
      for (int cg = 0; cg < 8; ++cg)
        o[cg] = wmma_bf16(aP1, bV1[cg], o[cg]);
    }
  }

  float inv[8];
#pragma unroll
  for (int r = 0; r < 8; ++r)
    inv[r] = 1.0f / __shfl(rsum, crow(lane, r));

#pragma unroll
  for (int cg = 0; cg < 8; ++cg) {
#pragma unroll
    for (int r = 0; r < 8; ++r) {
      const int qr = crow(lane, r);
      const size_t oidx = ((size_t)(b * N_ + qrow0 + qr)) * DH_ +
                          h * D_ + cg * 16 + (lane & 15);
      aout[oidx] = (bf16_t)(o[cg][r] * inv[r]);
    }
  }
}

// ---------------------------------------------------------------------------
// Kernel 4: projection GEMM. out = aout(16384x2048) @ proj_w^T + proj_b
// ---------------------------------------------------------------------------
__global__ void proj_gemm_kernel(const bf16_t* __restrict__ ain,
                                 const bf16_t* __restrict__ pw,
                                 const float* __restrict__ pb,
                                 float* __restrict__ out) {
  const int lane = threadIdx.x & 31;
  const int wave = threadIdx.x >> 5;
  const int row0 = blockIdx.x * 128 + wave * 32;
  const int col0 = blockIdx.y * 64;
  const int KT = DH_ / 32;    // 64

  const bf16_t* A = ain + (size_t)row0 * DH_;
  const bf16_t* W = pw + (size_t)col0 * DH_;
  v8f acc[2][4] = {};

  v16bf a0A, a1A, bA[4], a0B, a1B, bB[4];
  gemm_load(A, W, DH_, 0, lane, a0A, a1A, bA);
  for (int kt = 0; kt < KT; kt += 2) {
    gemm_load(A, W, DH_, kt + 1, lane, a0B, a1B, bB);
    if (kt + 4 < KT) {
      __builtin_prefetch(A + (kt + 4) * 32, 0, 1);
      __builtin_prefetch(W + (kt + 4) * 32, 0, 1);
    }
    gemm_mma(acc, a0A, a1A, bA);
    if (kt + 2 < KT) gemm_load(A, W, DH_, kt + 2, lane, a0A, a1A, bA);
    gemm_mma(acc, a0B, a1B, bB);
  }

#pragma unroll
  for (int m = 0; m < 2; ++m) {
    const int rowb = row0 + m * 16;
#pragma unroll
    for (int cg = 0; cg < 4; ++cg) {
      const int j = col0 + cg * 16 + (lane & 15);
      const float bj = pb[j];
#pragma unroll
      for (int r = 0; r < 8; ++r) {
        const int i = rowb + crow(lane, r);
        out[(size_t)i * DIM_ + j] = acc[m][cg][r] + bj;
      }
    }
  }
}

// ---------------------------------------------------------------------------
// Launcher
// ---------------------------------------------------------------------------
extern "C" void kernel_launch(void* const* d_in, const int* in_sizes, int n_in,
                              void* d_out, int out_size, void* d_ws, size_t ws_size,
                              hipStream_t stream) {
  const float* x      = (const float*)d_in[0];
  const float* gamma  = (const float*)d_in[1];
  const float* beta   = (const float*)d_in[2];
  const float* qkv_w  = (const float*)d_in[3];
  const float* qkv_b  = (const float*)d_in[4];
  const float* proj_w = (const float*)d_in[5];
  const float* proj_b = (const float*)d_in[6];
  const float* abias  = (const float*)d_in[7];
  const int*   bidx   = (const int*)d_in[8];
  float* out = (float*)d_out;

  char* wp = (char*)d_ws;
  auto carve = [&](size_t bytes) -> char* {
    char* p = wp;
    wp += (bytes + 255) & ~(size_t)255;
    return p;
  };
  bf16_t* xn = (bf16_t*)carve((size_t)B_ * N_ * DIM_ * 2);
  bf16_t* qw = (bf16_t*)carve((size_t)HQKV_ * DIM_ * 2);
  bf16_t* pw = (bf16_t*)carve((size_t)DIM_ * DH_ * 2);
  bf16_t* qb = (bf16_t*)carve((size_t)B_ * H_ * N_ * KD_ * 2);
  bf16_t* kb = (bf16_t*)carve((size_t)B_ * H_ * N_ * KD_ * 2);
  bf16_t* vT = (bf16_t*)carve((size_t)B_ * H_ * D_ * N_ * 2);
  bf16_t* aO = (bf16_t*)carve((size_t)B_ * N_ * DH_ * 2);

  {
    int n = HQKV_ * DIM_;
    cast_bf16_kernel<<<(n + 255) / 256, 256, 0, stream>>>(qkv_w, qw, n);
  }
  {
    int n = DIM_ * DH_;
    cast_bf16_kernel<<<(n + 255) / 256, 256, 0, stream>>>(proj_w, pw, n);
  }

  layernorm_kernel<<<(B_ * N_) / 8, 256, 0, stream>>>(x, gamma, beta, xn);

  {
    dim3 grid((B_ * N_) / 128, HQKV_ / 64);
    qkv_gemm_kernel<<<grid, 128, 0, stream>>>(xn, qw, qkv_b, qb, kb, vT);
  }

  {
    dim3 grid(N_ / 64, H_, B_);
    const size_t shmem = (size_t)KLDS_ELEMS * 2 +
                         (size_t)4 * 16 * PSTRIDE * 2;   // 81,920 + 133,120 B
    attn_kernel<<<grid, 128, shmem, stream>>>(qb, kb, vT, abias, bidx, aO);
  }

  {
    dim3 grid((B_ * N_) / 128, DIM_ / 64);
    proj_gemm_kernel<<<grid, 128, 0, stream>>>(aO, pw, proj_b, out);
  }
  (void)in_sizes; (void)n_in; (void)out_size; (void)ws_size;
}
